// BNTChannelAttention_4131758538762
// MI455X (gfx1250) — compile-verified, matrix operand order checked by reading, not run on hardware
//
#include <hip/hip_runtime.h>

// ---------------------------------------------------------------------------
// BNT channel attention for MI455X (gfx1250), wave32.
// Stage 1: qkv = x @ W_qkv + b   -- bf16x3 split-precision GEMM on
//          V_WMMA_F32_16X16X32_BF16 (3 matrix issues per 16x16x32 instead of
//          8 x V_WMMA_F32_16X16X4_F32; ~fp32 accuracy: err ~ sqrt(K)*2^-17)
// Stage 2: attn = softmax(q^T k / 8) per (b,h) -- exact f32 WMMA over N=4096
// Stage 3: out = V @ attn -- memory-bound, exact f32 WMMA; LDS tiles staged
//          by the Tensor Data Mover (tensor_load_to_lds + s_wait_tensorcnt)
//          with TDM padding reproducing the bank-conflict-free row stride.
// ---------------------------------------------------------------------------

typedef __attribute__((ext_vector_type(2)))  float   v2f;
typedef __attribute__((ext_vector_type(4)))  float   f4;
typedef __attribute__((ext_vector_type(8)))  float   v8f;
typedef __attribute__((ext_vector_type(4)))  __bf16  v4bf;
typedef __attribute__((ext_vector_type(8)))  __bf16  v8bf;
typedef __attribute__((ext_vector_type(16))) __bf16  v16bf;

typedef unsigned int u32;
typedef __attribute__((ext_vector_type(4))) u32 u32x4;
typedef __attribute__((ext_vector_type(4))) int i32x4;
typedef __attribute__((ext_vector_type(8))) int i32x8;

#define D_HID   1024
#define N_OUT   3072
#define SEQ     4096
#define BATCH   8
#define NH      16
#define HD      64
#define M_TOTAL (BATCH * SEQ)   // 32768

#if defined(__has_builtin)
#if __has_builtin(__builtin_amdgcn_tensor_load_to_lds) && \
    __has_builtin(__builtin_amdgcn_s_wait_tensorcnt)
#define USE_TDM 1
#endif
#endif

static __device__ __forceinline__ v8f wmma_f32(v2f a, v2f b, v8f c) {
    return __builtin_amdgcn_wmma_f32_16x16x4_f32(
        false, a, false, b, (short)0, c, false, false);
}
static __device__ __forceinline__ v8f wmma_bf16(v16bf a, v16bf b, v8f c) {
    return __builtin_amdgcn_wmma_f32_16x16x32_bf16(
        false, a, false, b, (short)0, c, false, false);
}

// split x into bf16 hi + bf16 lo (x ~= hi + lo, |err| ~ 2^-16 |x|)
static __device__ __forceinline__ void split4(f4 v, v4bf& h, v4bf& l) {
#pragma unroll
    for (int t = 0; t < 4; ++t) {
        float x = v[t];
        __bf16 hb = (__bf16)x;
        h[t] = hb;
        l[t] = (__bf16)(x - (float)hb);
    }
}

// 16-bit A/B fragment (16x32 / 32x16): lane's data = two contiguous 16B runs
// at K = lhi*8 and K = 16 + lhi*8 within the tile row.
static __device__ __forceinline__ v16bf load_frag(const __bf16* row, int lhi) {
    v8bf p0 = *(const v8bf*)(row + lhi * 8);
    v8bf p1 = *(const v8bf*)(row + 16 + lhi * 8);
    return __builtin_shufflevector(p0, p1, 0, 1, 2, 3, 4, 5, 6, 7,
                                           8, 9, 10, 11, 12, 13, 14, 15);
}

#ifdef USE_TDM
// Issue one 2D TDM tile load: global (strided rows) -> LDS (padded rows).
// dims/strides in 4-byte elements; pad codes per ISA D# group1 encoding.
static __device__ __forceinline__ void tdm_load_2d(
    u32 lds_byte_off, const void* gaddr,
    u32 tensor_d0, u32 tensor_d1, u32 stride0,
    u32 tile_d0, u32 tile_d1,
    u32 pad_interval_code, u32 pad_amount_code)
{
    unsigned long long ga = (unsigned long long)(size_t)gaddr;
    u32x4 g0 = { 1u,                                   // count=1, user mode
                 lds_byte_off,                         // lds_addr
                 (u32)ga,                              // global_addr[31:0]
                 (u32)((ga >> 32) & 0x1FFFFFFu) | (2u << 30) };  // addr[56:32] | type=2
    i32x8 g1;
    g1[0] = (int)((2u << 16) |                         // data_size = 4 bytes
                  (1u << 20) |                         // pad_enable
                  (pad_interval_code << 22) |
                  (pad_amount_code << 25));
    g1[1] = (int)((tensor_d0 & 0xFFFFu) << 16);        // [31:16]=tensor_dim0.lo
    g1[2] = (int)((tensor_d0 >> 16) | ((tensor_d1 & 0xFFFFu) << 16));
    g1[3] = (int)((tensor_d1 >> 16) | (tile_d0 << 16));
    g1[4] = (int)(tile_d1 & 0xFFFFu);                  // tile_dim1, tile_dim2=0
    g1[5] = (int)stride0;                              // tensor_dim0_stride[31:0]
    g1[6] = 0;
    g1[7] = 0;
    i32x4 z4 = { 0, 0, 0, 0 };
#if __clang_major__ >= 23
    i32x8 z8 = { 0, 0, 0, 0, 0, 0, 0, 0 };
    __builtin_amdgcn_tensor_load_to_lds(g0, g1, z4, z4, z8, 0);
#else
    __builtin_amdgcn_tensor_load_to_lds(g0, g1, z4, z4, 0);
#endif
}
#endif

// ============================ Stage 1: QKV GEMM =============================
// Block: 256 threads = 8 waves (2 x 4). Block tile 128x128, wave tile 64x32.
// K-slab KC=64 (two K=32 fragment steps per barrier).
#define BM  128
#define BN  128
#define KC  64
#define LDK 72   // bf16 row stride: 64 + 8 pad = 144 B (16B aligned, bank-spread)

__global__ __launch_bounds__(256)
void qkv_gemm_kernel(const float* __restrict__ X, const float* __restrict__ W,
                     const float* __restrict__ bias, float* __restrict__ Y)
{
    __shared__ __bf16 Ah[BM * LDK];   // x tile hi,  [m][k]
    __shared__ __bf16 Al[BM * LDK];   // x tile lo
    __shared__ __bf16 Bh[BN * LDK];   // W tile hi (transposed), [n][k]
    __shared__ __bf16 Bl[BN * LDK];   // W tile lo

    const int tid  = threadIdx.x;
    const int wave = tid >> 5;
    const int lane = tid & 31;
    const int l16  = lane & 15;
    const int lhi  = lane >> 4;

    const int nblk = N_OUT / BN;          // 24
    const int m0   = (blockIdx.x / nblk) * BM;
    const int n0   = (blockIdx.x % nblk) * BN;

    const int wm = wave >> 2;             // 0..1 : 64-row strip
    const int wn = wave & 3;              // 0..3 : 32-col strip

    v8f acc[4][2] = {};

    for (int k0 = 0; k0 < D_HID; k0 += KC) {
        // ---- stage X tile 128x64: 2048 float4 -> 8 per thread, split hi/lo
#pragma unroll
        for (int i = 0; i < 8; ++i) {
            int idx = tid + i * 256;
            int r   = idx >> 4;           // 0..127
            int c4  = idx & 15;           // 0..15
            f4 v = *(const f4*)(X + (size_t)(m0 + r) * D_HID + k0 + c4 * 4);
            v4bf h, l;
            split4(v, h, l);
            *(v4bf*)(&Ah[r * LDK + c4 * 4]) = h;
            *(v4bf*)(&Al[r * LDK + c4 * 4]) = l;
        }
        // ---- stage W tile 64x128 transposed -> Bs[n][k], split hi/lo
#pragma unroll
        for (int i = 0; i < 8; ++i) {
            int idx = tid + i * 256;
            int kr  = idx >> 5;           // 0..63
            int c4  = idx & 31;           // 0..31
            f4 v = *(const f4*)(W + (size_t)(k0 + kr) * N_OUT + n0 + c4 * 4);
#pragma unroll
            for (int t = 0; t < 4; ++t) {
                float  x  = v[t];
                __bf16 hb = (__bf16)x;
                Bh[(c4 * 4 + t) * LDK + kr] = hb;
                Bl[(c4 * 4 + t) * LDK + kr] = (__bf16)(x - (float)hb);
            }
        }
        __syncthreads();

        if (k0 + KC < D_HID) {   // pull next slab toward caches
            __builtin_prefetch(X + (size_t)(m0 + (tid >> 1)) * D_HID + k0 + KC, 0, 3);
            __builtin_prefetch(W + (size_t)(k0 + KC + (tid >> 2)) * N_OUT + n0, 0, 3);
        }

        // ---- compute: 2 K=32 steps, 24 WMMAs each (hi*hi + hi*lo + lo*hi)
#pragma unroll
        for (int kk = 0; kk < KC; kk += 32) {
            v16bf ah[4], al[4], bh[2], bl[2];
#pragma unroll
            for (int i = 0; i < 4; ++i) {
                const int row = (wm * 64 + i * 16 + l16) * LDK + kk;
                ah[i] = load_frag(&Ah[row], lhi);
                al[i] = load_frag(&Al[row], lhi);
            }
#pragma unroll
            for (int j = 0; j < 2; ++j) {
                const int row = (wn * 32 + j * 16 + l16) * LDK + kk;
                bh[j] = load_frag(&Bh[row], lhi);
                bl[j] = load_frag(&Bl[row], lhi);
            }
#pragma unroll
            for (int i = 0; i < 4; ++i)
#pragma unroll
                for (int j = 0; j < 2; ++j) {
                    acc[i][j] = wmma_bf16(ah[i], bh[j], acc[i][j]);
                    acc[i][j] = wmma_bf16(ah[i], bl[j], acc[i][j]);
                    acc[i][j] = wmma_bf16(al[i], bh[j], acc[i][j]);
                }
        }
        __syncthreads();
    }

    // ---- epilogue: + bias, store (lane-halves are 64B coalesced)
#pragma unroll
    for (int j = 0; j < 2; ++j) {
        const int col = n0 + wn * 32 + j * 16 + l16;
        const float bv = bias[col];
#pragma unroll
        for (int i = 0; i < 4; ++i) {
#pragma unroll
            for (int r = 0; r < 8; ++r) {
                int row = m0 + wm * 64 + i * 16 + r + lhi * 8;
                Y[(size_t)row * N_OUT + col] = acc[i][j][r] + bv;
            }
        }
    }
}

// ==================== Stage 2: S = q^T k / 8, softmax =======================
// Grid: 128 blocks (one per b,h). 128 threads = 4 waves in 2x2 over 64x64.
__global__ __launch_bounds__(128)
void attn_kernel(const float* __restrict__ QKV, float* __restrict__ ATTN)
{
    __shared__ float Smat[64 * 65];

    const int bh   = blockIdx.x;
    const int b    = bh >> 4;
    const int h    = bh & 15;
    const int tid  = threadIdx.x;
    const int wave = tid >> 5;
    const int lane = tid & 31;
    const int l16  = lane & 15;
    const int lhi  = lane >> 4;

    const int wm = wave >> 1;             // 0..1 : 32 d-rows
    const int wn = wave & 1;              // 0..1 : 32 e-cols

    const float* qbase = QKV + (size_t)b * SEQ * N_OUT + h * HD;   // q[n][d]
    const float* kbase = qbase + D_HID;                            // k[n][e]

    v8f acc[2][2] = {};

#pragma unroll 4
    for (int n = 0; n < SEQ; n += 4) {
        const int ka = n + lhi * 2;
        v2f a[2], bb[2];
#pragma unroll
        for (int i = 0; i < 2; ++i) {     // A[m=d, k=n] = q[n][d]
            const float* p = qbase + (size_t)ka * N_OUT + (wm * 32 + i * 16 + l16);
            v2f t; t.x = p[0]; t.y = p[N_OUT]; a[i] = t;
        }
#pragma unroll
        for (int j = 0; j < 2; ++j) {     // B[k=n, e] = k[n][e]
            const float* p = kbase + (size_t)ka * N_OUT + (wn * 32 + j * 16 + l16);
            v2f t; t.x = p[0]; t.y = p[N_OUT]; bb[j] = t;
        }
#pragma unroll
        for (int i = 0; i < 2; ++i)
#pragma unroll
            for (int j = 0; j < 2; ++j)
                acc[i][j] = wmma_f32(a[i], bb[j], acc[i][j]);
    }

    // scale by 1/sqrt(64) and park in LDS
#pragma unroll
    for (int i = 0; i < 2; ++i)
#pragma unroll
        for (int j = 0; j < 2; ++j)
#pragma unroll
            for (int r = 0; r < 8; ++r) {
                int row = wm * 32 + i * 16 + r + lhi * 8;
                int col = wn * 32 + j * 16 + l16;
                Smat[row * 65 + col] = acc[i][j][r] * 0.125f;
            }
    __syncthreads();

    // softmax over e (last axis), one row per thread
    if (tid < 64) {
        float mx = -3.402823466e+38f;
#pragma unroll 8
        for (int e = 0; e < 64; ++e) mx = fmaxf(mx, Smat[tid * 65 + e]);
        float s = 0.f;
#pragma unroll 8
        for (int e = 0; e < 64; ++e) {
            float ex = __expf(Smat[tid * 65 + e] - mx);
            Smat[tid * 65 + e] = ex;
            s += ex;
        }
        const float inv = 1.f / s;
        float* o = ATTN + (size_t)bh * (HD * HD);
#pragma unroll 8
        for (int e = 0; e < 64; ++e) o[tid * 64 + e] = Smat[tid * 65 + e] * inv;
    }
}

// ======================= Stage 3: out = V @ attn ============================
// Grid: 128 (b,h) x 32 row-tiles. 256 threads = 8 waves, each a 16x64 strip.
// V tile (128 x 64 f32, row stride 3072) and attn (64 x 64 f32) are staged
// into LDS by the Tensor Data Mover; TDM padding (64 dwords data + 4 dwords
// pad per row) reproduces the LDV=68 layout used by the WMMA fragment reads.
#define LDV 68   // 64 + 4 floats pad: 272 B row stride (16B-aligned)

__global__ __launch_bounds__(256)
void av_kernel(const float* __restrict__ QKV, const float* __restrict__ ATTN,
               float* __restrict__ OUT)
{
    __shared__ float Vs[128 * LDV];   // v tile  [n][d]
    __shared__ float Ps[64 * LDV];    // attn    [d][e]

    const int bh   = blockIdx.x >> 5;
    const int mt   = blockIdx.x & 31;
    const int b    = bh >> 4;
    const int h    = bh & 15;
    const int tid  = threadIdx.x;
    const int wave = tid >> 5;
    const int lane = tid & 31;
    const int l16  = lane & 15;
    const int lhi  = lane >> 4;
    const int n0   = mt * 128;

    const float* vbase = QKV + (size_t)b * SEQ * N_OUT + 2 * D_HID + h * HD; // v[n][d]
    const float* attn  = ATTN + (size_t)bh * (HD * HD);

#ifdef USE_TDM
    if (wave == 0) {
        // pad_interval code 5 -> every 64 dwords; pad_amount code 3 -> 4 dwords
        tdm_load_2d((u32)(size_t)(void*)Vs, vbase + (size_t)n0 * N_OUT,
                    /*tensor_d0=*/HD, /*tensor_d1=*/128, /*stride0=*/N_OUT,
                    /*tile_d0=*/HD, /*tile_d1=*/128, 5u, 3u);
        tdm_load_2d((u32)(size_t)(void*)Ps, attn,
                    /*tensor_d0=*/HD, /*tensor_d1=*/HD, /*stride0=*/HD,
                    /*tile_d0=*/HD, /*tile_d1=*/HD, 5u, 3u);
        __builtin_amdgcn_s_wait_tensorcnt(0);
    }
    __syncthreads();
#else
    // attn -> LDS (4 x float4 per thread)
#pragma unroll
    for (int i = 0; i < 4; ++i) {
        int idx = tid + i * 256;
        int r   = idx >> 4;               // d row 0..63
        int c4  = idx & 15;
        f4 v = *(const f4*)(attn + r * 64 + c4 * 4);
        *(f4*)(&Ps[r * LDV + c4 * 4]) = v;
    }
    // V tile -> LDS (8 x float4 per thread)
#pragma unroll
    for (int i = 0; i < 8; ++i) {
        int idx = tid + i * 256;
        int r   = idx >> 4;               // 0..127
        int c4  = idx & 15;
        f4 v = *(const f4*)(vbase + (size_t)(n0 + r) * N_OUT + c4 * 4);
        *(f4*)(&Vs[r * LDV + c4 * 4]) = v;
    }
    __syncthreads();
#endif

    v8f acc[4] = {};
#pragma unroll
    for (int k = 0; k < HD; k += 4) {
        const int ka = k + lhi * 2;
        v2f a = *(const v2f*)(&Vs[(wave * 16 + l16) * LDV + ka]);
#pragma unroll
        for (int j = 0; j < 4; ++j) {
            v2f bb;
            bb.x = Ps[(ka + 0) * LDV + j * 16 + l16];
            bb.y = Ps[(ka + 1) * LDV + j * 16 + l16];
            acc[j] = wmma_f32(a, bb, acc[j]);
        }
    }

    // out[b, n, h*64 + e]
#pragma unroll
    for (int j = 0; j < 4; ++j)
#pragma unroll
        for (int r = 0; r < 8; ++r) {
            int row = n0 + wave * 16 + r + lhi * 8;
            int col = j * 16 + l16;
            OUT[((size_t)b * SEQ + row) * D_HID + h * HD + col] = acc[j][r];
        }
}

// ================================ launch ====================================
extern "C" void kernel_launch(void* const* d_in, const int* in_sizes, int n_in,
                              void* d_out, int out_size, void* d_ws, size_t ws_size,
                              hipStream_t stream)
{
    const float* x    = (const float*)d_in[0];   // [8, 4096, 1024]
    const float* W    = (const float*)d_in[1];   // [1024, 3072]
    const float* bias = (const float*)d_in[2];   // [3072]
    float* out = (float*)d_out;                  // [8, 4096, 1024]

    float* qkv  = (float*)d_ws;                          // 32768 x 3072 f32
    float* attn = qkv + (size_t)M_TOTAL * N_OUT;         // 128 x 64 x 64 f32

    qkv_gemm_kernel<<<(M_TOTAL / BM) * (N_OUT / BN), 256, 0, stream>>>(x, W, bias, qkv);
    attn_kernel<<<BATCH * NH, 128, 0, stream>>>(qkv, attn);
    av_kernel<<<BATCH * NH * (SEQ / 128), 256, 0, stream>>>(qkv, attn, out);
}
